// MultiLatentAttentionOptimized_16312285790940
// MI455X (gfx1250) — matrix operand order repaired and checked by reading, not compiled, not run on hardware
//
#include <hip/hip_runtime.h>
#include <hip/hip_bf16.h>

typedef __bf16 bf16;
typedef __attribute__((ext_vector_type(16))) __bf16 v16bf;
typedef __attribute__((ext_vector_type(8)))  __bf16 v8bf;
typedef __attribute__((ext_vector_type(8)))  float  v8f;

#define DIM 1024
#define SEQ 2048
#define NB  2
#define NH  8
#define MROWS (NB*SEQ)          // 4096
#define QC 128
#define QNOPE 96
#define QROPE 32
#define KVC 128
#define KNOPE 64
#define KROPE 64
#define VHD 256
#define HD 128                   // attention head dim

// ---------------------------------------------------------------------------
// CDNA5 async memory->LDS DMA (ASYNCcnt-tracked, no VGPR transit)
// ---------------------------------------------------------------------------
__device__ inline void async_load_b128(const void* gaddr, const void* lds_ptr) {
    unsigned lds = (unsigned)(unsigned long long)lds_ptr;   // low 32 bits = LDS byte offset
    asm volatile("global_load_async_to_lds_b128 %0, %1, off"
                 :: "v"(lds), "v"(gaddr) : "memory");
}
__device__ inline void wait_async0() {
    asm volatile("s_wait_asynccnt 0x0" ::: "memory");
}

// ---------------------------------------------------------------------------
// WMMA helpers (CDNA5 16x16x32 bf16, f32 accumulate)
// ---------------------------------------------------------------------------
__device__ inline v16bf frag_join(v8bf lo, v8bf hi) {
    v16bf r;
#pragma unroll
    for (int i = 0; i < 8; ++i) { r[i] = lo[i]; r[i + 8] = hi[i]; }
    return r;
}

// A fragment (16x32, row-major, row stride = ld elems).
// lane m=lane&15, half=lane>>4; elem e<8 -> K=half*8+e, e>=8 -> K=16+half*8+(e-8)
__device__ inline v16bf load_a_frag(const bf16* p, int ld) {
    int lane = threadIdx.x & 31;
    int m = lane & 15, half = lane >> 4;
    const bf16* row = p + m * ld;
    v8bf lo = *(const v8bf*)(row + half * 8);
    v8bf hi = *(const v8bf*)(row + 16 + half * 8);
    return frag_join(lo, hi);
}

// B fragment (32x16) from K-contiguous storage: p -> [n=0][k=0], n-rows stride ld.
// lane n=lane&15, kh=lane>>4; elem e<8 -> K=kh*8+e, e>=8 -> K=16+kh*8+(e-8)
__device__ inline v16bf load_bT_frag(const bf16* p, int ld) {
    int lane = threadIdx.x & 31;
    int n = lane & 15, kh = lane >> 4;
    const bf16* row = p + n * ld + kh * 8;
    v8bf lo = *(const v8bf*)(row);
    v8bf hi = *(const v8bf*)(row + 16);
    return frag_join(lo, hi);
}

__device__ inline v8f wmma_bf16(v16bf a, v16bf b, v8f c) {
    return __builtin_amdgcn_wmma_f32_16x16x32_bf16(false, a, false, b, (short)0, c, false, false);
}

// ---------------------------------------------------------------------------
// f32 -> bf16 conversion (straight, and transposing for weights: [K][N] -> [N][K])
// ---------------------------------------------------------------------------
__global__ void cvt_f32_bf16_kernel(const float* __restrict__ in, bf16* __restrict__ out, int n) {
    int i = blockIdx.x * 256 + threadIdx.x;
    if (i < n) out[i] = (bf16)in[i];
}

__global__ void cvt_t_f32_bf16_kernel(const float* __restrict__ in, bf16* __restrict__ out,
                                      int K, int N) {
    int idx = blockIdx.x * 256 + threadIdx.x;
    if (idx >= K * N) return;
    int n = idx % N;
    int k = idx / N;                       // coalesced read along N
    out[(size_t)n * K + k] = (bf16)in[idx];
}

#define BK 32
#define AS_LD 48   // 96B rows -> 16B aligned fragment loads
#define BT_LD 48

// ---------------------------------------------------------------------------
// Big-tile WMMA GEMM: 128x128 tile, 256 threads (8 waves, 4x2), BK=32,
// double-buffered async LDS staging. Requires M%128==0 && N%128==0.
// Each wave: 32x64 -> 8 WMMAs per K-step.
// ---------------------------------------------------------------------------
__global__ __launch_bounds__(256) void gemm_bf16_128_kernel(
    const bf16* __restrict__ A, const bf16* __restrict__ Bt,
    const float* __restrict__ bias, float* __restrict__ C,
    int M, int N, int K)
{
    __shared__ bf16 As[2][128 * AS_LD];   // 24 KB
    __shared__ bf16 Bs[2][128 * BT_LD];   // 24 KB

    int tid  = threadIdx.x;
    int wave = tid >> 5;
    int lane = tid & 31;
    int row0 = blockIdx.y * 128;
    int col0 = blockIdx.x * 128;
    int m_off = (wave >> 1) * 32;         // 4 wave-rows
    int n_off = (wave & 1) * 64;          // 2 wave-cols

    v8f acc[2][4] = {};

    // each tile: 128 rows x 32 cols = 512 x 16B transfers, 2/thread/operand
    auto issue_tile = [&](int buf, int k0) {
#pragma unroll
        for (int it = 0; it < 2; ++it) {
            int i = tid + it * 256;
            int r = i >> 2;
            int c = (i & 3) * 8;
            async_load_b128(&A[(size_t)(row0 + r) * K + k0 + c], &As[buf][r * AS_LD + c]);
        }
#pragma unroll
        for (int it = 0; it < 2; ++it) {
            int i = tid + it * 256;
            int r = i >> 2;
            int c = (i & 3) * 8;
            async_load_b128(&Bt[(size_t)(col0 + r) * K + k0 + c], &Bs[buf][r * BT_LD + c]);
        }
    };

    issue_tile(0, 0);
    int nk = K / BK;
    for (int t = 0; t < nk; ++t) {
        int buf = t & 1;
        wait_async0();
        __syncthreads();
        if (t + 1 < nk) issue_tile(buf ^ 1, (t + 1) * BK);

        v16bf a0 = load_a_frag(&As[buf][(m_off + 0)  * AS_LD], AS_LD);
        v16bf a1 = load_a_frag(&As[buf][(m_off + 16) * AS_LD], AS_LD);
        v16bf b0 = load_bT_frag(&Bs[buf][(n_off + 0)  * BT_LD], BT_LD);
        v16bf b1 = load_bT_frag(&Bs[buf][(n_off + 16) * BT_LD], BT_LD);
        v16bf b2 = load_bT_frag(&Bs[buf][(n_off + 32) * BT_LD], BT_LD);
        v16bf b3 = load_bT_frag(&Bs[buf][(n_off + 48) * BT_LD], BT_LD);

        acc[0][0] = wmma_bf16(a0, b0, acc[0][0]);
        acc[0][1] = wmma_bf16(a0, b1, acc[0][1]);
        acc[0][2] = wmma_bf16(a0, b2, acc[0][2]);
        acc[0][3] = wmma_bf16(a0, b3, acc[0][3]);
        acc[1][0] = wmma_bf16(a1, b0, acc[1][0]);
        acc[1][1] = wmma_bf16(a1, b1, acc[1][1]);
        acc[1][2] = wmma_bf16(a1, b2, acc[1][2]);
        acc[1][3] = wmma_bf16(a1, b3, acc[1][3]);
        __syncthreads();
    }

    int n_in = lane & 15;
    int mh   = (lane >> 4) * 8;
#pragma unroll
    for (int i = 0; i < 2; ++i)
#pragma unroll
        for (int j = 0; j < 4; ++j) {
            int col = col0 + n_off + j * 16 + n_in;
            float bv = bias ? bias[col] : 0.f;
#pragma unroll
            for (int r = 0; r < 8; ++r) {
                int rr = row0 + m_off + i * 16 + mh + r;
                C[(size_t)rr * N + col] = acc[i][j][r] + bv;
            }
        }
}

// ---------------------------------------------------------------------------
// Small-tile WMMA GEMM: 64x64 tile, 128 threads (4 waves), used when N%128!=0.
// ---------------------------------------------------------------------------
__global__ __launch_bounds__(128) void gemm_bf16_64_kernel(
    const bf16* __restrict__ A, const bf16* __restrict__ Bt,
    const float* __restrict__ bias, float* __restrict__ C,
    int M, int N, int K)
{
    __shared__ bf16 As[2][64 * AS_LD];
    __shared__ bf16 Bs[2][64 * BT_LD];

    int tid  = threadIdx.x;
    int wave = tid >> 5;
    int lane = tid & 31;
    int row0 = blockIdx.y * 64;
    int col0 = blockIdx.x * 64;
    int wr = (wave >> 1) * 32;
    int wc = (wave & 1) * 32;

    v8f acc[2][2] = {};

    auto issue_tile = [&](int buf, int k0) {
#pragma unroll
        for (int it = 0; it < 2; ++it) {
            int i = tid + it * 128;
            int r = i >> 2;
            int c = (i & 3) * 8;
            async_load_b128(&A[(size_t)(row0 + r) * K + k0 + c], &As[buf][r * AS_LD + c]);
        }
#pragma unroll
        for (int it = 0; it < 2; ++it) {
            int i = tid + it * 128;
            int r = i >> 2;
            int c = (i & 3) * 8;
            async_load_b128(&Bt[(size_t)(col0 + r) * K + k0 + c], &Bs[buf][r * BT_LD + c]);
        }
    };

    issue_tile(0, 0);
    int nk = K / BK;
    for (int t = 0; t < nk; ++t) {
        int buf = t & 1;
        wait_async0();
        __syncthreads();
        if (t + 1 < nk) issue_tile(buf ^ 1, (t + 1) * BK);

        v16bf a0 = load_a_frag(&As[buf][(wr + 0)  * AS_LD], AS_LD);
        v16bf a1 = load_a_frag(&As[buf][(wr + 16) * AS_LD], AS_LD);
        v16bf b0 = load_bT_frag(&Bs[buf][(wc + 0)  * BT_LD], BT_LD);
        v16bf b1 = load_bT_frag(&Bs[buf][(wc + 16) * BT_LD], BT_LD);

        acc[0][0] = wmma_bf16(a0, b0, acc[0][0]);
        acc[0][1] = wmma_bf16(a0, b1, acc[0][1]);
        acc[1][0] = wmma_bf16(a1, b0, acc[1][0]);
        acc[1][1] = wmma_bf16(a1, b1, acc[1][1]);
        __syncthreads();
    }

    int n_in = lane & 15;
    int mh   = (lane >> 4) * 8;
#pragma unroll
    for (int i = 0; i < 2; ++i)
#pragma unroll
        for (int j = 0; j < 2; ++j) {
            int col = col0 + wc + j * 16 + n_in;
            float bv = bias ? bias[col] : 0.f;
#pragma unroll
            for (int r = 0; r < 8; ++r) {
                int rr = row0 + wr + i * 16 + mh + r;
                C[(size_t)rr * N + col] = acc[i][j][r] + bv;
            }
        }
}

// ---------------------------------------------------------------------------
// RMSNorm epilogue: in f32 [rows, ld] (first 128 cols) -> out bf16 [rows,128]
// ---------------------------------------------------------------------------
__global__ __launch_bounds__(128) void rmsnorm_bf16_kernel(
    const float* __restrict__ in, int ld, const float* __restrict__ w,
    bf16* __restrict__ out)
{
    int row = blockIdx.x;
    int c = threadIdx.x;
    float v = in[(size_t)row * ld + c];
    float ss = v * v;
#pragma unroll
    for (int m = 16; m >= 1; m >>= 1) ss += __shfl_xor(ss, m, 32);
    __shared__ float part[4];
    if ((c & 31) == 0) part[c >> 5] = ss;
    __syncthreads();
    float tot = part[0] + part[1] + part[2] + part[3];
    float r = rsqrtf(tot * (1.0f / 128.0f) + 1e-8f);
    out[(size_t)row * 128 + c] = (bf16)(w[c] * v * r);
}

// ---------------------------------------------------------------------------
// RoPE on shared k_rope: tmp_kv cols [128,192) -> bf16 [rows, 64]
// ---------------------------------------------------------------------------
__global__ void rope_k_kernel(const float* __restrict__ tkv, const int* __restrict__ pos_ids,
                              bf16* __restrict__ kr, int total)
{
    int idx = blockIdx.x * 256 + threadIdx.x;
    if (idx >= total) return;
    int row = idx >> 5;
    int i = idx & 31;
    float inv = __powf(10000.f, -(2.f * (float)i) / 64.f);
    float ang = (float)pos_ids[row] * inv;
    float sn, cs;
    __sincosf(ang, &sn, &cs);
    const float* src = tkv + (size_t)row * (KVC + KROPE) + KVC;
    float xe = src[2 * i], xo = src[2 * i + 1];
    kr[(size_t)row * 64 + 2 * i]     = (bf16)(xe * cs - xo * sn);
    kr[(size_t)row * 64 + 2 * i + 1] = (bf16)(xe * sn + xo * cs);
}

// ---------------------------------------------------------------------------
// Scatter q (f32 [rows,1024]) -> q_states bf16 [B,H,S,128] with RoPE on last 32 dims
// ---------------------------------------------------------------------------
__global__ void q_scatter_kernel(const float* __restrict__ q, const int* __restrict__ pos_ids,
                                 bf16* __restrict__ qs, int total)
{
    int idx = blockIdx.x * 256 + threadIdx.x;
    if (idx >= total) return;
    int d = idx & 127;
    int h = (idx >> 7) & 7;
    int row = idx >> 10;
    int b = row >> 11;
    int s = row & (SEQ - 1);
    const float* qr = q + (size_t)row * (NH * HD);
    float val;
    if (d < QNOPE) {
        val = qr[h * QNOPE + d];
    } else {
        int dd = d - QNOPE;
        int i = dd >> 1;
        float inv = __powf(10000.f, -(2.f * (float)i) / 32.f);
        float ang = (float)pos_ids[row] * inv;
        float sn, cs;
        __sincosf(ang, &sn, &cs);
        const float* rp = qr + NH * QNOPE + h * QROPE;
        float xe = rp[2 * i], xo = rp[2 * i + 1];
        val = (dd & 1) ? (xe * sn + xo * cs) : (xe * cs - xo * sn);
    }
    qs[(((size_t)(b * NH + h)) * SEQ + s) * HD + d] = (bf16)val;
}

// ---------------------------------------------------------------------------
// Scatter kv (f32 [rows,2560]) + krope ->
//   k_states bf16 [B,H,S,128]  and  vT_states bf16 [B,H,256,S]  (pre-transposed!)
// ---------------------------------------------------------------------------
__global__ void kv_scatter_kernel(const float* __restrict__ kv, const bf16* __restrict__ krope,
                                  bf16* __restrict__ ks, bf16* __restrict__ vt, int total)
{
    int idx = blockIdx.x * 256 + threadIdx.x;
    if (idx >= total) return;
    int d = idx % 384;
    int t = idx / 384;
    int h = t & 7;
    int row = t >> 3;
    int b = row >> 11;
    int s = row & (SEQ - 1);
    size_t bh = (size_t)(b * NH + h);
    if (d < KNOPE) {
        ks[(bh * SEQ + s) * HD + d] = (bf16)kv[(size_t)row * 2560 + h * KNOPE + d];
    } else if (d < HD) {
        ks[(bh * SEQ + s) * HD + d] = krope[(size_t)row * 64 + (d - KNOPE)];
    } else {
        int vd = d - HD;
        vt[(bh * VHD + vd) * SEQ + s] = (bf16)kv[(size_t)row * 2560 + NH * KNOPE + h * VHD + vd];
    }
}

// ---------------------------------------------------------------------------
// Flash attention: per (b,h), 64 query rows per WG (4 waves x 16 rows).
// K block [64][128] and Vt block [256][64] staged via async DMA; online softmax;
// all matmuls via WMMA. Output attn bf16 [B, S, H*256].
// ---------------------------------------------------------------------------
#define KS_LD 136   // 272B rows, 16B aligned
#define VT_LD 72    // 144B rows
#define PS_LD 72

__global__ __launch_bounds__(128) void mla_attn_kernel(
    const bf16* __restrict__ Q, const bf16* __restrict__ Kst, const bf16* __restrict__ VTst,
    bf16* __restrict__ O, float scale)
{
    __shared__ bf16 Ks[64 * KS_LD];        // 17,408 B
    __shared__ bf16 Vt[VHD * VT_LD];       // 36,864 B  ([vcol][key])
    __shared__ bf16 Ps[4 * 16 * PS_LD];    //  9,216 B  (wave-private P tiles)

    int tid = threadIdx.x, wave = tid >> 5, lane = tid & 31;
    int bh = blockIdx.y;
    int q0 = blockIdx.x * 64 + wave * 16;

    const bf16* Qb  = Q    + (size_t)bh * SEQ * HD;
    const bf16* Kb  = Kst  + (size_t)bh * SEQ * HD;
    const bf16* VTb = VTst + (size_t)bh * VHD * SEQ;

    v16bf qf[4];
#pragma unroll
    for (int kk = 0; kk < 4; ++kk)
        qf[kk] = load_a_frag(Qb + (size_t)q0 * HD + kk * 32, HD);

    v8f o[16];
#pragma unroll
    for (int i = 0; i < 16; ++i) o[i] = (v8f)0.f;
    float m_r[8], l_r[8];
#pragma unroll
    for (int r = 0; r < 8; ++r) { m_r[r] = -3.0e38f; l_r[r] = 0.f; }

    bf16* myPs = &Ps[wave * 16 * PS_LD];

    for (int kb = 0; kb < SEQ; kb += 64) {
        // stage K block [64][128]: 1024 x 16B transfers, 8/thread
#pragma unroll
        for (int it = 0; it < 8; ++it) {
            int i = tid + it * 128;
            int r = i >> 4;
            int c = (i & 15) * 8;
            async_load_b128(&Kb[(size_t)(kb + r) * HD + c], &Ks[r * KS_LD + c]);
        }
        // stage Vt block [256][64]: 2048 x 16B transfers, 16/thread
#pragma unroll
        for (int it = 0; it < 16; ++it) {
            int i = tid + it * 128;
            int r = i >> 3;
            int c = (i & 7) * 8;
            async_load_b128(&VTb[(size_t)r * SEQ + kb + c], &Vt[r * VT_LD + c]);
        }
        wait_async0();
        __syncthreads();

        // scores S[16,64] = Q * K^T
        v8f sc[4];
#pragma unroll
        for (int nt = 0; nt < 4; ++nt) sc[nt] = (v8f)0.f;
#pragma unroll
        for (int nt = 0; nt < 4; ++nt)
#pragma unroll
            for (int kk = 0; kk < 4; ++kk) {
                v16bf bfr = load_bT_frag(&Ks[(nt * 16) * KS_LD + kk * 32], KS_LD);
                sc[nt] = wmma_bf16(qf[kk], bfr, sc[nt]);
            }

        // online softmax (rows per half-wave; reduce across 16 lanes)
#pragma unroll
        for (int r = 0; r < 8; ++r) {
            float mx = -3.0e38f;
#pragma unroll
            for (int nt = 0; nt < 4; ++nt) {
                sc[nt][r] *= scale;
                mx = fmaxf(mx, sc[nt][r]);
            }
#pragma unroll
            for (int m = 8; m >= 1; m >>= 1) mx = fmaxf(mx, __shfl_xor(mx, m, 32));
            float mn = fmaxf(m_r[r], mx);
            float corr = __expf(m_r[r] - mn);
            m_r[r] = mn;
            float rs = 0.f;
#pragma unroll
            for (int nt = 0; nt < 4; ++nt) {
                float p = __expf(sc[nt][r] - mn);
                sc[nt][r] = p;
                rs += p;
            }
#pragma unroll
            for (int m = 8; m >= 1; m >>= 1) rs += __shfl_xor(rs, m, 32);
            l_r[r] = l_r[r] * corr + rs;
#pragma unroll
            for (int nt2 = 0; nt2 < 16; ++nt2) o[nt2][r] *= corr;
        }

        // P (C layout) -> A fragments via wave-private LDS
        int half = lane >> 4, n_in = lane & 15;
#pragma unroll
        for (int nt = 0; nt < 4; ++nt)
#pragma unroll
            for (int r = 0; r < 8; ++r)
                myPs[(r + 8 * half) * PS_LD + nt * 16 + n_in] = (bf16)sc[nt][r];

        v16bf pf0 = load_a_frag(myPs,      PS_LD);
        v16bf pf1 = load_a_frag(myPs + 32, PS_LD);

        // O += P * V
#pragma unroll
        for (int nt2 = 0; nt2 < 16; ++nt2) {
            v16bf v0 = load_bT_frag(&Vt[(nt2 * 16) * VT_LD + 0],  VT_LD);
            v16bf v1 = load_bT_frag(&Vt[(nt2 * 16) * VT_LD + 32], VT_LD);
            o[nt2] = wmma_bf16(pf0, v0, o[nt2]);
            o[nt2] = wmma_bf16(pf1, v1, o[nt2]);
        }
        __syncthreads();
    }

    // normalize and store to attn [B, S, H*256]
    int b = bh >> 3, h = bh & 7;
    int half = lane >> 4, n_in = lane & 15;
#pragma unroll
    for (int nt2 = 0; nt2 < 16; ++nt2)
#pragma unroll
        for (int r = 0; r < 8; ++r) {
            int s = q0 + r + 8 * half;
            float val = o[nt2][r] / l_r[r];
            O[((size_t)b * SEQ + s) * (NH * VHD) + h * VHD + nt2 * 16 + n_in] = (bf16)val;
        }
}

// ---------------------------------------------------------------------------
// Host launcher
// ---------------------------------------------------------------------------
extern "C" void kernel_launch(void* const* d_in, const int* in_sizes, int n_in,
                              void* d_out, int out_size, void* d_ws, size_t ws_size,
                              hipStream_t stream) {
    const float* x         = (const float*)d_in[0];
    const int*   pos       = (const int*)  d_in[1];
    const float* w_dq_w    = (const float*)d_in[2];
    const float* w_dq_b    = (const float*)d_in[3];
    const float* q_norm_w  = (const float*)d_in[4];
    const float* w_uq_w    = (const float*)d_in[5];
    const float* w_uq_b    = (const float*)d_in[6];
    const float* w_dkv_w   = (const float*)d_in[7];
    const float* w_dkv_b   = (const float*)d_in[8];
    const float* kv_norm_w = (const float*)d_in[9];
    const float* w_ukuv_w  = (const float*)d_in[10];
    const float* w_ukuv_b  = (const float*)d_in[11];
    const float* w_o_w     = (const float*)d_in[12];
    const float* w_o_b     = (const float*)d_in[13];
    float* out = (float*)d_out;

    char* wsb = (char*)d_ws;
    size_t off = 0;
    auto take = [&](size_t bytes) -> char* {
        char* p = wsb + off;
        off += (bytes + 255) & ~(size_t)255;
        return p;
    };

    bf16* x_bf       = (bf16*)take((size_t)MROWS * DIM * 2);
    bf16* w_dq_bt    = (bf16*)take((size_t)DIM * QC * 2);                  // [128][1024]
    bf16* w_dkv_bt   = (bf16*)take((size_t)DIM * (KVC + KROPE) * 2);       // [192][1024]
    bf16* w_uq_bt    = (bf16*)take((size_t)QC * (NH * HD) * 2);            // [1024][128]
    bf16* w_ukuv_bt  = (bf16*)take((size_t)KVC * (NH * (KNOPE + VHD)) * 2);// [2560][128]
    bf16* w_o_bt     = (bf16*)take((size_t)(NH * VHD) * DIM * 2);          // [1024][2048]
    float* tmp_q     = (float*)take((size_t)MROWS * QC * 4);
    float* tmp_kv    = (float*)take((size_t)MROWS * (KVC + KROPE) * 4);
    bf16* cq_bf      = (bf16*)take((size_t)MROWS * QC * 2);
    bf16* ckv_bf     = (bf16*)take((size_t)MROWS * KVC * 2);
    bf16* krope_bf   = (bf16*)take((size_t)MROWS * KROPE * 2);
    float* qfull     = (float*)take((size_t)MROWS * (NH * HD) * 4);
    float* kvfull    = (float*)take((size_t)MROWS * (NH * (KNOPE + VHD)) * 4);
    bf16* q_states   = (bf16*)take((size_t)NB * NH * SEQ * HD * 2);
    bf16* k_states   = (bf16*)take((size_t)NB * NH * SEQ * HD * 2);
    bf16* vT_states  = (bf16*)take((size_t)NB * NH * VHD * SEQ * 2);
    bf16* attn_bf    = (bf16*)take((size_t)MROWS * (NH * VHD) * 2);

    // x straight; weights transposed to [N][K] for async-friendly GEMM staging
    cvt_f32_bf16_kernel<<<(MROWS * DIM + 255) / 256, 256, 0, stream>>>(x, x_bf, MROWS * DIM);
    auto cvt_t = [&](const float* src, bf16* dst, int K, int N) {
        int n = K * N;
        cvt_t_f32_bf16_kernel<<<(n + 255) / 256, 256, 0, stream>>>(src, dst, K, N);
    };
    cvt_t(w_dq_w,   w_dq_bt,   DIM, QC);
    cvt_t(w_dkv_w,  w_dkv_bt,  DIM, KVC + KROPE);
    cvt_t(w_uq_w,   w_uq_bt,   QC,  NH * HD);
    cvt_t(w_ukuv_w, w_ukuv_bt, KVC, NH * (KNOPE + VHD));
    cvt_t(w_o_w,    w_o_bt,    NH * VHD, DIM);

    // big-tile GEMM for N%128==0, small-tile fallback otherwise
    auto gemm = [&](const bf16* A, const bf16* Bt, const float* bias, float* C,
                    int M, int N, int K) {
        if ((N & 127) == 0) {
            dim3 grid(N / 128, M / 128);
            gemm_bf16_128_kernel<<<grid, 256, 0, stream>>>(A, Bt, bias, C, M, N, K);
        } else {
            dim3 grid(N / 64, M / 64);
            gemm_bf16_64_kernel<<<grid, 128, 0, stream>>>(A, Bt, bias, C, M, N, K);
        }
    };

    // down projections
    gemm(x_bf, w_dq_bt,  w_dq_b,  tmp_q,  MROWS, QC,          DIM);
    gemm(x_bf, w_dkv_bt, w_dkv_b, tmp_kv, MROWS, KVC + KROPE, DIM);

    // norms + shared-k rope
    rmsnorm_bf16_kernel<<<MROWS, 128, 0, stream>>>(tmp_q,  QC,          q_norm_w,  cq_bf);
    rmsnorm_bf16_kernel<<<MROWS, 128, 0, stream>>>(tmp_kv, KVC + KROPE, kv_norm_w, ckv_bf);
    {
        int total = MROWS * 32;
        rope_k_kernel<<<(total + 255) / 256, 256, 0, stream>>>(tmp_kv, pos, krope_bf, total);
    }

    // up projections
    gemm(cq_bf,  w_uq_bt,   w_uq_b,   qfull,  MROWS, NH * HD,            QC);
    gemm(ckv_bf, w_ukuv_bt, w_ukuv_b, kvfull, MROWS, NH * (KNOPE + VHD), KVC);

    // head scatters (+q rope); v written pre-transposed [B,H,256,S]
    {
        int total = MROWS * NH * HD;
        q_scatter_kernel<<<(total + 255) / 256, 256, 0, stream>>>(qfull, pos, q_states, total);
    }
    {
        int total = MROWS * NH * (HD + VHD);
        kv_scatter_kernel<<<(total + 255) / 256, 256, 0, stream>>>(kvfull, krope_bf, k_states, vT_states, total);
    }

    // attention
    {
        dim3 grid(SEQ / 64, NB * NH);
        float scale = 1.0f / sqrtf((float)HD);
        mla_attn_kernel<<<grid, 128, 0, stream>>>(q_states, k_states, vT_states, attn_bf, scale);
    }

    // output projection -> d_out (f32)
    gemm(attn_bf, w_o_bt, w_o_b, out, MROWS, DIM, NH * VHD);
}
